// Decoder_86002425135288
// MI455X (gfx1250) — matrix operand order repaired and checked by reading, not compiled, not run on hardware
//
#include <hip/hip_runtime.h>
#include <hip/hip_bf16.h>
#include <math.h>

// ---------------------------------------------------------------------------
// Types for CDNA5 WMMA (wave32): bf16 16x16x32 -> f32 accumulate
// ---------------------------------------------------------------------------
typedef __bf16 bf16_t;
typedef bf16_t v16bf __attribute__((ext_vector_type(16)));
typedef float  v8f   __attribute__((ext_vector_type(8)));

union FragBits {
    uint4 q[2];   // 32 bytes == 16 bf16
    v16bf v;
};

__device__ __forceinline__ uint32_t f2bfbits(float f) {
    union { float f; uint32_t u; } x; x.f = f;
    uint32_t u = x.u;
    u += 0x7FFFu + ((u >> 16) & 1u);   // round-to-nearest-even
    return u >> 16;
}
__device__ __forceinline__ uint32_t pack2bf(float lo, float hi) {
    return f2bfbits(lo) | (f2bfbits(hi) << 16);
}
__device__ __forceinline__ float sigmoidf(float x) {
    return 1.0f / (1.0f + expf(-x));
}

// Problem constants
#define BB   128
#define HH   1024
#define EE   512
#define KX   3584          // [e(512) | out(1024) | h(1024)]
#define LDX  3584
#define LDY  2048

// ---------------------------------------------------------------------------
// fp32 -> bf16 conversion (vectorized, once per launch for the weights)
// ---------------------------------------------------------------------------
__global__ void cvt_bf16_kernel(const float* __restrict__ src,
                                uint16_t* __restrict__ dst, int n4)
{
    const int i = blockIdx.x * blockDim.x + threadIdx.x;
    if (i < n4) {
        const float4 f = reinterpret_cast<const float4*>(src)[i];
        uint2 o;
        o.x = pack2bf(f.x, f.y);
        o.y = pack2bf(f.z, f.w);
        reinterpret_cast<uint2*>(dst)[i] = o;
    }
}

// ---------------------------------------------------------------------------
// Software-pipelined K loop over one weight segment.  All fragment loads use
// compile-time immediate offsets (LDW is a template constant); only two
// address pointers (A row, W row) are live, each advanced 64 elems/iter.
// The K loop is kept rolled (#pragma unroll 1) so the ping-pong register
// double-buffering survives and waits stay partial.
//   kLen must be a multiple of 64.
// ---------------------------------------------------------------------------
template <int NT, int LDW>
__device__ __forceinline__ void gemm_seg(const uint16_t* __restrict__ aCur,
                                         const uint16_t* __restrict__ wCur,
                                         int kLen, v8f (&acc)[NT])
{
    FragBits afA, afB;
    FragBits bfA[NT], bfB[NT];

    // prologue: chunk 0 -> buffer A
    afA.q[0] = *reinterpret_cast<const uint4*>(aCur);
    afA.q[1] = *reinterpret_cast<const uint4*>(aCur + 16);
#pragma unroll
    for (int j = 0; j < NT; ++j) {
        bfA[j].q[0] = *reinterpret_cast<const uint4*>(wCur + j * 16 * LDW);
        bfA[j].q[1] = *reinterpret_cast<const uint4*>(wCur + j * 16 * LDW + 16);
    }

#pragma unroll 1
    for (int k0 = 0; k0 < kLen; k0 += 64) {
        // prefetch chunk k0+32 -> buffer B (imm offsets from current base)
        afB.q[0] = *reinterpret_cast<const uint4*>(aCur + 32);
        afB.q[1] = *reinterpret_cast<const uint4*>(aCur + 48);
#pragma unroll
        for (int j = 0; j < NT; ++j) {
            bfB[j].q[0] = *reinterpret_cast<const uint4*>(wCur + j * 16 * LDW + 32);
            bfB[j].q[1] = *reinterpret_cast<const uint4*>(wCur + j * 16 * LDW + 48);
        }

#pragma unroll
        for (int j = 0; j < NT; ++j)
            acc[j] = __builtin_amdgcn_wmma_f32_16x16x32_bf16(
                false, afA.v, false, bfA[j].v, (short)0, acc[j], false, false);

        if (k0 + 64 < kLen) {        // prefetch chunk k0+64 -> buffer A
            afA.q[0] = *reinterpret_cast<const uint4*>(aCur + 64);
            afA.q[1] = *reinterpret_cast<const uint4*>(aCur + 80);
#pragma unroll
            for (int j = 0; j < NT; ++j) {
                bfA[j].q[0] = *reinterpret_cast<const uint4*>(wCur + j * 16 * LDW + 64);
                bfA[j].q[1] = *reinterpret_cast<const uint4*>(wCur + j * 16 * LDW + 80);
            }
        }

#pragma unroll
        for (int j = 0; j < NT; ++j)
            acc[j] = __builtin_amdgcn_wmma_f32_16x16x32_bf16(
                false, afB.v, false, bfB[j].v, (short)0, acc[j], false, false);

        aCur += 64;
        wCur += 64;
    }
}

// ---------------------------------------------------------------------------
// Thin GEMM: C[128, N] = A_bf16[128, K] x W_bf16[N, K]^T  (WMMA bf16)
//   K = KS (+ K2): k in [0,KS) -> W1 (row stride LDW1);
//                  k in [KS,KS+K2) -> W2 (row stride LDW2)
//   MODE 0: +bias1[n]+bias2[n], store f32          (LSTM gates)
//   MODE 1: store f32                              (attention query)
//   MODE 2: tanh, store f32 AND bf16 copy          (decoder output)
// Block = 256 threads = 8 waves arranged 2(M) x 4(N); each wave owns NT
// 16x16 tiles along N.  Block tile: 32(M) x 64*NT(N).
// ---------------------------------------------------------------------------
template <int NT, int KS, int LDW1, int K2, int LDW2, int MODE>
__global__ void gemm_bf16_kernel(
    const uint16_t* __restrict__ A, int lda,
    const uint16_t* __restrict__ W1,
    const uint16_t* __restrict__ W2,
    float* __restrict__ outF, int ldc,
    const float* __restrict__ bias1, const float* __restrict__ bias2,
    uint16_t* __restrict__ outB, int ldc2)
{
    const int lane = threadIdx.x & 31;
    const int wid  = threadIdx.x >> 5;
    const int wm   = wid >> 2;          // 0..1
    const int wn   = wid & 3;           // 0..3
    const int mTile = blockIdx.y * 32 + wm * 16;
    const int nBase = blockIdx.x * (64 * NT) + wn * 16 * NT;

    v8f acc[NT];
#pragma unroll
    for (int j = 0; j < NT; ++j)
#pragma unroll
        for (int r = 0; r < 8; ++r) acc[j][r] = 0.0f;

    // A fragment (16-bit A 16x32, ISA 7.12.2):
    //  lanes 0-15: row M=lane, K octets {0..7, 16..23}
    //  lanes 16-31: row M=lane-16, K octets {8..15, 24..31}
    const int aoct = (lane >> 4) << 3;          // 0 or 8
    const uint16_t* aRow = A + (size_t)(mTile + (lane & 15)) * lda + aoct;

    // B fragment (32x16 bf16): lane n = lane&15; same K octets as A:
    // lanes<16 hold K {0..7, 16..23}, lanes>=16 hold K {8..15, 24..31}.
    // => lane reads two 8-bf16 octets from W[ncol][kh..kh+7], [kh+16..kh+23];
    //    tile j adds the compile-time offset j*16*LDW.
    const int khoff = (lane >> 4) << 3;         // 0 or 8
    const int ncol0 = nBase + (lane & 15);

    gemm_seg<NT, LDW1>(aRow, W1 + (size_t)ncol0 * LDW1 + khoff, KS, acc);
    if constexpr (K2 > 0)
        gemm_seg<NT, LDW2>(aRow + KS, W2 + (size_t)ncol0 * LDW2 + khoff, K2, acc);

    // C layout (32-bit C/D 16x16): lanes 0-15 -> N=lane, M = r (VGPR r);
    // lanes 16-31 -> N=lane-16, M = 8 + r.
    const int nLane = lane & 15;
    const int mBase = mTile + ((lane >> 4) << 3);
#pragma unroll
    for (int j = 0; j < NT; ++j) {
        const int n = nBase + 16 * j + nLane;
        if constexpr (MODE == 0) {
            const float bsum = bias1[n] + bias2[n];
#pragma unroll
            for (int r = 0; r < 8; ++r)
                outF[(size_t)(mBase + r) * ldc + n] = acc[j][r] + bsum;
        } else if constexpr (MODE == 1) {
#pragma unroll
            for (int r = 0; r < 8; ++r)
                outF[(size_t)(mBase + r) * ldc + n] = acc[j][r];
        } else {
#pragma unroll
            for (int r = 0; r < 8; ++r) {
                const float v = tanhf(acc[j][r]);
                outF[(size_t)(mBase + r) * ldc + n] = v;
                outB[(size_t)(mBase + r) * ldc2 + n] = (uint16_t)f2bfbits(v);
            }
        }
    }
}

// ---------------------------------------------------------------------------
// Initial state: h0 = concat(enc_h[0], enc_h[1]) (bf16 into X), c0, out0 = 0
// ---------------------------------------------------------------------------
__global__ void init_kernel(const float* __restrict__ enc_h,
                            const float* __restrict__ enc_c,
                            uint16_t* __restrict__ X,
                            float* __restrict__ c)
{
    const int idx = blockIdx.x * blockDim.x + threadIdx.x;   // < 131072
    const int b = idx >> 10, d = idx & 1023;
    const size_t lo = (size_t)b * 512 + (d & 511);
    const size_t off = (d < 512) ? lo : lo + (size_t)BB * 512;
    X[(size_t)b * LDX + 1536 + d] = (uint16_t)f2bfbits(enc_h[off]);
    X[(size_t)b * LDX + 512 + d]  = 0;                        // out0 = 0
    c[idx] = enc_c[off];
}

// ---------------------------------------------------------------------------
// Embedding gather for step t -> bf16 into X[:, 0:512]
// ---------------------------------------------------------------------------
__global__ void pack_e_kernel(const float* __restrict__ emb,
                              const int* __restrict__ trg_t,
                              uint16_t* __restrict__ X)
{
    const int idx = blockIdx.x * blockDim.x + threadIdx.x;   // < 65536
    const int b = idx >> 9, e = idx & 511;
    const float v = emb[(size_t)trg_t[b] * EE + e];
    X[(size_t)b * LDX + e] = (uint16_t)f2bfbits(v);
}

// ---------------------------------------------------------------------------
// LSTM pointwise: c,h update; h(bf16) -> X h-slot (next step) and Y[:,1024:]
// ---------------------------------------------------------------------------
__global__ void lstm_kernel(const float* __restrict__ gates,
                            float* __restrict__ c,
                            uint16_t* __restrict__ X,
                            uint16_t* __restrict__ Y)
{
    const int idx = blockIdx.x * blockDim.x + threadIdx.x;   // < 131072
    const int b = idx >> 10, d = idx & 1023;
    const float* g = gates + (size_t)b * 4096;
    const float ig = sigmoidf(g[d]);
    const float fg = sigmoidf(g[1024 + d]);
    const float gg = tanhf(g[2048 + d]);
    const float og = sigmoidf(g[3072 + d]);
    const float cn = fg * c[idx] + ig * gg;
    const float hn = og * tanhf(cn);
    c[idx] = cn;
    const uint16_t hb = (uint16_t)f2bfbits(hn);
    X[(size_t)b * LDX + 1536 + d] = hb;
    Y[(size_t)b * LDY + 1024 + d] = hb;
}

// ---------------------------------------------------------------------------
// Attention per batch element b: scores -> softmax -> context vector (bf16)
// One block (8 wave32) per b.
// ---------------------------------------------------------------------------
__global__ void attn_kernel(const float* __restrict__ q,
                            const float* __restrict__ ctx,
                            const unsigned char* __restrict__ mask,
                            uint16_t* __restrict__ Y)
{
    const int b = blockIdx.x;
    const int tid = threadIdx.x;
    const int lane = tid & 31, wid = tid >> 5;
    __shared__ float qs[HH];
    __shared__ float sc[128];
    __shared__ float red[128];

    for (int i = tid; i < HH; i += 256) qs[i] = q[(size_t)b * HH + i];
    __syncthreads();

    // scores: wave wid handles s = wid*16 .. wid*16+15 ; dot over H=1024
#pragma unroll 1
    for (int i = 0; i < 16; ++i) {
        const int s = wid * 16 + i;
        const float* cp = ctx + (size_t)s * (BB * HH) + (size_t)b * HH;
        float p = 0.0f;
        for (int j = lane; j < HH; j += 32) p += qs[j] * cp[j];
#pragma unroll
        for (int off = 16; off; off >>= 1) p += __shfl_xor(p, off, 32);
        if (lane == 0)
            sc[s] = mask[(size_t)s * BB + b] ? p : -INFINITY;
    }
    __syncthreads();

    // softmax over S=128
    if (tid < 128) red[tid] = sc[tid];
    __syncthreads();
    for (int off = 64; off; off >>= 1) {
        if (tid < off) red[tid] = fmaxf(red[tid], red[tid + off]);
        __syncthreads();
    }
    const float mx = red[0];
    __syncthreads();
    if (tid < 128) { const float e = expf(sc[tid] - mx); sc[tid] = e; red[tid] = e; }
    __syncthreads();
    for (int off = 64; off; off >>= 1) {
        if (tid < off) red[tid] += red[tid + off];
        __syncthreads();
    }
    const float inv = 1.0f / red[0];
    __syncthreads();
    if (tid < 128) sc[tid] *= inv;
    __syncthreads();

    // context vector cv[d] = sum_s a[s]*ctx[b,s,d]  -> bf16 into Y[:,0:1024]
    for (int d = tid; d < HH; d += 256) {
        const float* cp = ctx + (size_t)b * HH + d;
        float a = 0.0f;
#pragma unroll 4
        for (int s = 0; s < 128; ++s) a += sc[s] * cp[(size_t)s * (BB * HH)];
        Y[(size_t)b * LDY + d] = (uint16_t)f2bfbits(a);
    }
}

// ---------------------------------------------------------------------------
// Launch
// ---------------------------------------------------------------------------
extern "C" void kernel_launch(void* const* d_in, const int* in_sizes, int n_in,
                              void* d_out, int out_size, void* d_ws, size_t ws_size,
                              hipStream_t stream)
{
    const float*         ctx   = (const float*)d_in[0];          // [S,B,H]
    const unsigned char* mask  = (const unsigned char*)d_in[1];  // [S,B] bool
    const int*           trg   = (const int*)d_in[2];            // [T,B]
    const float*         enc_h = (const float*)d_in[3];          // [2,B,D]
    const float*         enc_c = (const float*)d_in[4];          // [2,B,D]
    const float*         emb   = (const float*)d_in[5];          // [VOCAB,E]
    const float*         Win   = (const float*)d_in[6];          // [H,H]
    const float*         Wout  = (const float*)d_in[7];          // [H,2H]
    const float*         Wih   = (const float*)d_in[8];          // [4H,E+H]
    const float*         Whh   = (const float*)d_in[9];          // [4H,H]
    const float*         bih   = (const float*)d_in[10];         // [4H]
    const float*         bhh   = (const float*)d_in[11];         // [4H]
    float* out = (float*)d_out;                                  // [T-1,B,H]

    char* p = (char*)d_ws;
    uint16_t* X      = (uint16_t*)p; p += (size_t)BB * KX  * 2;      // [128,3584]
    uint16_t* Y      = (uint16_t*)p; p += (size_t)BB * LDY * 2;      // [128,2048]
    float*    gates  = (float*)p;    p += (size_t)BB * 4096 * 4;     // [128,4096]
    float*    cbuf   = (float*)p;    p += (size_t)BB * HH * 4;       // [128,1024]
    float*    qbuf   = (float*)p;    p += (size_t)BB * HH * 4;       // [128,1024]
    uint16_t* WihB   = (uint16_t*)p; p += (size_t)4096 * 1536 * 2;   // bf16 weights
    uint16_t* WhhB   = (uint16_t*)p; p += (size_t)4096 * 1024 * 2;
    uint16_t* WinB   = (uint16_t*)p; p += (size_t)1024 * 1024 * 2;
    uint16_t* WoutB  = (uint16_t*)p; p += (size_t)1024 * 2048 * 2;
    (void)ws_size; (void)in_sizes; (void)n_in; (void)out_size;

    // One-time (per launch) weight conversion to bf16
    {
        const int n1 = 4096 * 1536 / 4, n2 = 4096 * 1024 / 4;
        const int n3 = 1024 * 1024 / 4, n4 = 1024 * 2048 / 4;
        cvt_bf16_kernel<<<(n1 + 255) / 256, 256, 0, stream>>>(Wih,  WihB,  n1);
        cvt_bf16_kernel<<<(n2 + 255) / 256, 256, 0, stream>>>(Whh,  WhhB,  n2);
        cvt_bf16_kernel<<<(n3 + 255) / 256, 256, 0, stream>>>(Win,  WinB,  n3);
        cvt_bf16_kernel<<<(n4 + 255) / 256, 256, 0, stream>>>(Wout, WoutB, n4);
    }

    init_kernel<<<512, 256, 0, stream>>>(enc_h, enc_c, X, cbuf);

    for (int t = 0; t < 127; ++t) {
        // x_t = [emb(trg[t]), out_{t-1}] ; X h-slot holds h_{t-1}
        pack_e_kernel<<<256, 256, 0, stream>>>(emb, trg + (size_t)t * BB, X);

        // gates = x @ Wih^T + h @ Whh^T + bih + bhh   (M=128,N=4096,K=1536+2048)
        gemm_bf16_kernel<4, 1536, 1536, 2048, 1024, 0>
            <<<dim3(16, 4), 256, 0, stream>>>(
                X, LDX, WihB, WhhB, gates, 4096, bih, bhh, nullptr, 0);

        lstm_kernel<<<512, 256, 0, stream>>>(gates, cbuf, X, Y);

        // q = h @ Win^T   (M=128,N=1024,K=1024)
        gemm_bf16_kernel<2, 1024, 1024, 0, 1024, 1>
            <<<dim3(8, 4), 256, 0, stream>>>(
                X + 1536, LDX, WinB, nullptr, qbuf, 1024, nullptr, nullptr,
                nullptr, 0);

        attn_kernel<<<128, 256, 0, stream>>>(qbuf, ctx, mask, Y);

        // out_t = tanh([cv,h] @ Wout^T)  (M=128,N=1024,K=2048)
        // -> f32 slice of d_out and bf16 into X out-slot for step t+1
        gemm_bf16_kernel<2, 2048, 2048, 0, 1024, 2>
            <<<dim3(8, 4), 256, 0, stream>>>(
                Y, LDY, WoutB, nullptr, out + (size_t)t * (BB * HH), 1024,
                nullptr, nullptr, X + 512, LDX);
    }
}
